// MSDeformAttn_30958124270101
// MI455X (gfx1250) — matrix operand order repaired and checked by reading, not compiled, not run on hardware
//
#include <hip/hip_runtime.h>
#include <hip/hip_bf16.h>
#include <math.h>

typedef __bf16 bf16_t;
typedef __attribute__((ext_vector_type(16))) __bf16 v16bf;
typedef __attribute__((ext_vector_type(8)))  __bf16 v8bf;
typedef __attribute__((ext_vector_type(8)))  float  v8f;

#define LEN    16660
#define DM     256
#define NH     8
#define HD     32
#define KDIM   256
#define NDIM   256

// ---------------------------------------------------------------------------
// f32 -> bf16 elementwise convert
// ---------------------------------------------------------------------------
__global__ void f32_to_bf16_kernel(const float* __restrict__ src,
                                   bf16_t* __restrict__ dst, int n) {
  int i = blockIdx.x * blockDim.x + threadIdx.x;
  if (i < n) dst[i] = (bf16_t)src[i];
}

// ---------------------------------------------------------------------------
// Weight transpose + convert: Wt[n*256 + k] = (bf16) W[k*256 + n]
// (makes per-lane B-fragment loads contiguous 32B)
// ---------------------------------------------------------------------------
__global__ void wtrans_kernel(const float* __restrict__ W,
                              bf16_t* __restrict__ Wt) {
  int i = blockIdx.x * blockDim.x + threadIdx.x;   // 65536 threads
  int n = i >> 8;
  int k = i & 255;
  Wt[(size_t)n * 256 + k] = (bf16_t)W[(size_t)k * 256 + n];
}

// ---------------------------------------------------------------------------
// bf16 WMMA GEMM:  C[M x 256] = A[M x 256] * B[256 x 256] + bias
//   A      : row-major bf16
//   Bt     : B transposed, i.e. Bt[n*256 + k] (bf16)
//   Cf/Cb  : optional f32 / bf16 outputs (row-major M x 256)
// One wave -> one 16x64 strip (4 N-tiles): A fragment loaded once per K-step
// and reused across 4 WMMAs (4x less A traffic from L2).
// ---------------------------------------------------------------------------
__global__ void gemm_bf16_wmma_kernel(const bf16_t* __restrict__ A,
                                      const bf16_t* __restrict__ Bt,
                                      const float*  __restrict__ bias,
                                      float*        __restrict__ Cf,
                                      bf16_t*       __restrict__ Cb,
                                      int M) {
  const int NGRP = NDIM / 64;                // 4 strips of 64 columns
  int wave = (blockIdx.x * blockDim.x + threadIdx.x) >> 5;
  int lane = threadIdx.x & 31;
  int mt = wave / NGRP;
  int ng = wave % NGRP;
  int mbase = mt * 16;
  if (mbase >= M) return;                    // wave-uniform

  int lh = lane & 15;                        // row (A) / col (B,C)
  int kh = lane >> 4;                        // K-half selector
  int arow = mbase + lh;
  if (arow >= M) arow = M - 1;               // clamp for safe reads (store guarded)

  const bf16_t* Aptr = A + (size_t)arow * KDIM;
  const bf16_t* Bptr0 = Bt + (size_t)(ng * 64 + 0 * 16 + lh) * KDIM;
  const bf16_t* Bptr1 = Bt + (size_t)(ng * 64 + 1 * 16 + lh) * KDIM;
  const bf16_t* Bptr2 = Bt + (size_t)(ng * 64 + 2 * 16 + lh) * KDIM;
  const bf16_t* Bptr3 = Bt + (size_t)(ng * 64 + 3 * 16 + lh) * KDIM;

  v8f acc0 = {}, acc1 = {}, acc2 = {}, acc3 = {};
#pragma unroll
  for (int k0 = 0; k0 < KDIM; k0 += 32) {
    // A fragment (16x32 bf16): lane(0-15) holds K = 0..15 +base,
    // lane(16-31) holds K = 16..31 +base -> two contiguous v8bf loads.
    v8bf alo = *(const v8bf*)(Aptr + k0 + kh * 16);
    v8bf ahi = *(const v8bf*)(Aptr + k0 + kh * 16 + 8);
    v16bf a;
#pragma unroll
    for (int i = 0; i < 8; ++i) { a[i] = alo[i]; a[8 + i] = ahi[i]; }

    // B fragments (32x16 bf16): lanes 0-15 hold K=k0..k0+15 for their column,
    // lanes 16-31 hold K=k0+16..k0+31 -> one contiguous 32B v16bf load each.
    v16bf b0 = *(const v16bf*)(Bptr0 + k0 + kh * 16);
    v16bf b1 = *(const v16bf*)(Bptr1 + k0 + kh * 16);
    v16bf b2 = *(const v16bf*)(Bptr2 + k0 + kh * 16);
    v16bf b3 = *(const v16bf*)(Bptr3 + k0 + kh * 16);

    acc0 = __builtin_amdgcn_wmma_f32_16x16x32_bf16(false, a, false, b0, (short)0, acc0, false, false);
    acc1 = __builtin_amdgcn_wmma_f32_16x16x32_bf16(false, a, false, b1, (short)0, acc1, false, false);
    acc2 = __builtin_amdgcn_wmma_f32_16x16x32_bf16(false, a, false, b2, (short)0, acc2, false, false);
    acc3 = __builtin_amdgcn_wmma_f32_16x16x32_bf16(false, a, false, b3, (short)0, acc3, false, false);
  }

#pragma unroll
  for (int j = 0; j < 4; ++j) {
    v8f acc = (j == 0) ? acc0 : (j == 1) ? acc1 : (j == 2) ? acc2 : acc3;
    int bcol = ng * 64 + j * 16 + lh;
    float bb = bias ? bias[bcol] : 0.0f;
#pragma unroll
    for (int r = 0; r < 8; ++r) {
      int grow = mbase + kh * 8 + r;         // D: VGPR r -> row r + 8*(lane/16)
      if (grow < M) {
        float val = acc[r] + bb;
        size_t off = (size_t)grow * NDIM + bcol;
        if (Cf) Cf[off] = val;
        if (Cb) Cb[off] = (bf16_t)val;
      }
    }
  }
}

// ---------------------------------------------------------------------------
// Deformable sampling + per-point attention.
// One thread per (query, head). Two passes over the 16 (level,point) samples:
//   pass 1: bilinear-sample K, logits = q . k / sqrt(32); softmax over 16
//   pass 2: bilinear-sample V, weighted accumulate; write bf16 h
// ---------------------------------------------------------------------------
__device__ __forceinline__ float corner_dot(const float* __restrict__ kf,
                                            const float qv[HD],
                                            int yi, int xi, int Hl, int Wl,
                                            int base, int h, float w) {
  if (xi < 0 || xi >= Wl || yi < 0 || yi >= Hl) return 0.0f;
  const float4* kp = (const float4*)(kf + ((size_t)(base + yi * Wl + xi)) * DM + h * HD);
  float d = 0.0f;
#pragma unroll
  for (int j = 0; j < 8; ++j) {
    float4 kk = kp[j];
    d += qv[4*j+0]*kk.x + qv[4*j+1]*kk.y + qv[4*j+2]*kk.z + qv[4*j+3]*kk.w;
  }
  return w * d;
}

__device__ __forceinline__ void corner_acc(const float* __restrict__ vf,
                                           float out[HD],
                                           int yi, int xi, int Hl, int Wl,
                                           int base, int h, float w) {
  if (xi < 0 || xi >= Wl || yi < 0 || yi >= Hl) return;
  const float4* vp = (const float4*)(vf + ((size_t)(base + yi * Wl + xi)) * DM + h * HD);
#pragma unroll
  for (int j = 0; j < 8; ++j) {
    float4 vv = vp[j];
    out[4*j+0] += w * vv.x; out[4*j+1] += w * vv.y;
    out[4*j+2] += w * vv.z; out[4*j+3] += w * vv.w;
  }
}

__global__ void deform_attn_kernel(const float* __restrict__ q,
                                   const float* __restrict__ kf,
                                   const float* __restrict__ vf,
                                   const float* __restrict__ offs,
                                   const float* __restrict__ refp,
                                   bf16_t* __restrict__ hout) {
  int t = blockIdx.x * blockDim.x + threadIdx.x;
  if (t >= LEN * NH) return;
  int qi = t >> 3;
  int h  = t & 7;

  const int Hs[4] = {112, 56, 28, 14};
  const int Ws[4] = {112, 56, 28, 14};
  const int Ls[4] = {0, 12544, 15680, 16464};

  float qv[HD];
  const float* qp = q + (size_t)qi * DM + h * HD;
#pragma unroll
  for (int d = 0; d < HD; ++d) qv[d] = qp[d];

  const float* op = offs + (size_t)qi * DM + h * HD;   // [l*8 + p*2 + c]
  float logits[16];

  // ---- pass 1: key sampling -> logits ----
#pragma unroll
  for (int l = 0; l < 4; ++l) {
    float rx = refp[(size_t)qi * 8 + l * 2 + 0];
    float ry = refp[(size_t)qi * 8 + l * 2 + 1];
    int Hl = Hs[l], Wl = Ws[l], base = Ls[l];
#pragma unroll
    for (int p = 0; p < 4; ++p) {
      float ox = op[l * 8 + p * 2 + 0];
      float oy = op[l * 8 + p * 2 + 1];
      float x = (rx + ox / (float)Wl) * (float)Wl - 0.5f;
      float y = (ry + oy / (float)Hl) * (float)Hl - 0.5f;
      float xf = floorf(x), yf = floorf(y);
      float lx = x - xf,  ly = y - yf;
      int x0 = (int)xf,   y0 = (int)yf;
      float a = 0.0f;
      a += corner_dot(kf, qv, y0,     x0,     Hl, Wl, base, h, (1.f-ly)*(1.f-lx));
      a += corner_dot(kf, qv, y0,     x0 + 1, Hl, Wl, base, h, (1.f-ly)*lx);
      a += corner_dot(kf, qv, y0 + 1, x0,     Hl, Wl, base, h, ly*(1.f-lx));
      a += corner_dot(kf, qv, y0 + 1, x0 + 1, Hl, Wl, base, h, ly*lx);
      logits[l * 4 + p] = a * 0.17677669529663687f;    // 1/sqrt(32)
    }
  }

  // ---- softmax over 16 points ----
  float mx = logits[0];
#pragma unroll
  for (int i = 1; i < 16; ++i) mx = fmaxf(mx, logits[i]);
  float s = 0.0f;
#pragma unroll
  for (int i = 0; i < 16; ++i) { logits[i] = __expf(logits[i] - mx); s += logits[i]; }
  float inv = 1.0f / s;

  // ---- pass 2: value sampling, weighted accumulate ----
  float out[HD];
#pragma unroll
  for (int d = 0; d < HD; ++d) out[d] = 0.0f;

#pragma unroll
  for (int l = 0; l < 4; ++l) {
    float rx = refp[(size_t)qi * 8 + l * 2 + 0];
    float ry = refp[(size_t)qi * 8 + l * 2 + 1];
    int Hl = Hs[l], Wl = Ws[l], base = Ls[l];
#pragma unroll
    for (int p = 0; p < 4; ++p) {
      float ox = op[l * 8 + p * 2 + 0];
      float oy = op[l * 8 + p * 2 + 1];
      float x = (rx + ox / (float)Wl) * (float)Wl - 0.5f;
      float y = (ry + oy / (float)Hl) * (float)Hl - 0.5f;
      float xf = floorf(x), yf = floorf(y);
      float lx = x - xf,  ly = y - yf;
      int x0 = (int)xf,   y0 = (int)yf;
      float aw = logits[l * 4 + p] * inv;
      corner_acc(vf, out, y0,     x0,     Hl, Wl, base, h, aw * (1.f-ly)*(1.f-lx));
      corner_acc(vf, out, y0,     x0 + 1, Hl, Wl, base, h, aw * (1.f-ly)*lx);
      corner_acc(vf, out, y0 + 1, x0,     Hl, Wl, base, h, aw * ly*(1.f-lx));
      corner_acc(vf, out, y0 + 1, x0 + 1, Hl, Wl, base, h, aw * ly*lx);
    }
  }

  bf16_t* hp = hout + (size_t)qi * DM + h * HD;
#pragma unroll
  for (int d = 0; d < HD; ++d) hp[d] = (bf16_t)out[d];
}

// ---------------------------------------------------------------------------
// Host launcher
// ---------------------------------------------------------------------------
extern "C" void kernel_launch(void* const* d_in, const int* in_sizes, int n_in,
                              void* d_out, int out_size, void* d_ws, size_t ws_size,
                              hipStream_t stream) {
  const float* query = (const float*)d_in[0];
  const float* refp  = (const float*)d_in[1];
  const float* xin   = (const float*)d_in[2];
  // d_in[3]/d_in[4] (spatial_shapes / level_start_index) are compile-time fixed.
  const float* W_q   = (const float*)d_in[5];
  const float* b_q   = (const float*)d_in[6];
  const float* W_k   = (const float*)d_in[7];
  const float* b_k   = (const float*)d_in[8];
  const float* W_v   = (const float*)d_in[9];
  const float* b_v   = (const float*)d_in[10];
  const float* W_o   = (const float*)d_in[11];
  const float* b_o   = (const float*)d_in[12];
  const float* W_off = (const float*)d_in[13];
  const float* b_off = (const float*)d_in[14];
  float* out = (float*)d_out;

  char* ws = (char*)d_ws;
  size_t pos = 0;
  auto alloc = [&](size_t bytes) -> void* {
    void* r = (void*)(ws + pos);
    pos += (bytes + 255) & ~(size_t)255;
    return r;
  };

  const size_t MN = (size_t)LEN * DM;
  float*  qf       = (float*)alloc(MN * 4);
  float*  kfb      = (float*)alloc(MN * 4);
  float*  vfb      = (float*)alloc(MN * 4);
  float*  offs     = (float*)alloc(MN * 4);
  bf16_t* query_bf = (bf16_t*)alloc(MN * 2);
  bf16_t* x_bf     = (bf16_t*)alloc(MN * 2);
  bf16_t* q_bf     = (bf16_t*)alloc(MN * 2);
  bf16_t* h_bf     = (bf16_t*)alloc(MN * 2);
  bf16_t* Wq_t     = (bf16_t*)alloc(65536 * 2);
  bf16_t* Wk_t     = (bf16_t*)alloc(65536 * 2);
  bf16_t* Wv_t     = (bf16_t*)alloc(65536 * 2);
  bf16_t* Wo_t     = (bf16_t*)alloc(65536 * 2);
  bf16_t* Woff_t   = (bf16_t*)alloc(65536 * 2);

  // activations -> bf16
  int nconv = (int)MN;
  int cblocks = (nconv + 255) / 256;
  f32_to_bf16_kernel<<<cblocks, 256, 0, stream>>>(query, query_bf, nconv);
  f32_to_bf16_kernel<<<cblocks, 256, 0, stream>>>(xin,   x_bf,     nconv);

  // weights -> transposed bf16
  wtrans_kernel<<<256, 256, 0, stream>>>(W_q,   Wq_t);
  wtrans_kernel<<<256, 256, 0, stream>>>(W_k,   Wk_t);
  wtrans_kernel<<<256, 256, 0, stream>>>(W_v,   Wv_t);
  wtrans_kernel<<<256, 256, 0, stream>>>(W_o,   Wo_t);
  wtrans_kernel<<<256, 256, 0, stream>>>(W_off, Woff_t);

  // WMMA projection GEMMs: each wave computes a 16x64 strip (4 tiles)
  const int mtiles = (LEN + 15) / 16;          // 1042
  const int waves  = mtiles * (NDIM / 64);     // 4168
  const int gblocks = (waves * 32 + 255) / 256; // 521
  gemm_bf16_wmma_kernel<<<gblocks, 256, 0, stream>>>(query_bf, Wq_t,   b_q,   qf,   q_bf,    LEN);
  gemm_bf16_wmma_kernel<<<gblocks, 256, 0, stream>>>(x_bf,     Wk_t,   b_k,   kfb,  nullptr, LEN);
  gemm_bf16_wmma_kernel<<<gblocks, 256, 0, stream>>>(x_bf,     Wv_t,   b_v,   vfb,  nullptr, LEN);
  gemm_bf16_wmma_kernel<<<gblocks, 256, 0, stream>>>(q_bf,     Woff_t, b_off, offs, nullptr, LEN);

  // sampling + attention
  int nthr = LEN * NH;
  deform_attn_kernel<<<(nthr + 255) / 256, 256, 0, stream>>>(qf, kfb, vfb, offs, refp, h_bf);

  // output projection directly into d_out
  gemm_bf16_wmma_kernel<<<gblocks, 256, 0, stream>>>(h_bf, Wo_t, b_o, out, nullptr, LEN);
}
